// SGC_45397804319028
// MI455X (gfx1250) — matrix operand order repaired and checked by reading, not compiled, not run on hardware
//
#include <hip/hip_runtime.h>
#include <hip/hip_bf16.h>

#define N_NODES 8192
#define NFEAT   256
#define NCLASS  40
#define SCALE   1.0f

typedef __bf16 v16bf __attribute__((ext_vector_type(16)));
typedef float  v8f   __attribute__((ext_vector_type(8)));
typedef unsigned int u32x4 __attribute__((ext_vector_type(4)));

union Frag {            // 32 bytes: one WMMA bf16 A/B operand
    v16bf v;
    u32x4 q[2];
};

// ---------------------------------------------------------------------------
// Fused ContraNorm: out = (1+SCALE)*x - SCALE * softmax(qn @ qn^T) @ x
// qn = row-normalized x (bf16), vv = x (bf16). Flash-style: logits are dots of
// unit vectors in [-1,1], so exp() needs no max-subtraction -> single pass.
// Block = 256 threads = 8 waves; wave pair (grp) owns 16 rows; `half` splits
// the S n-tiles (keys) and the O column tiles between the two waves of a pair.
// ---------------------------------------------------------------------------
__global__ void __launch_bounds__(256)
contranorm_flash(const float* __restrict__ xin,
                 const __bf16* __restrict__ qn,
                 const __bf16* __restrict__ vv,
                 float* __restrict__ out)
{
    __shared__ alignas(16) __bf16 sKV[256 * 72];   // phase1: K [64][256]; phase2: V^T [256][72]
    __shared__ alignas(16) __bf16 sP[4][16 * 72];  // staged P per row-group: [16 rows][64(+pad)]
    __shared__ float sRS[64][2];                   // softmax denominator halves

    const int tid  = threadIdx.x;
    const int lane = tid & 31;
    const int wave = tid >> 5;
    const int grp  = wave >> 1;          // row group 0..3 (16 rows each)
    const int half = wave & 1;           // split within the pair
    const int hi   = lane >> 4;          // lane half (WMMA layouts)
    const int l16  = lane & 15;
    const int rowbase = blockIdx.x * 64 + grp * 16;

    // Q fragments in A layout, K dim = 8 chunks of 32 (resident for whole sweep)
    v16bf qf[8];
    {
        const __bf16* qp = qn + (size_t)(rowbase + l16) * NFEAT;
#pragma unroll
        for (int c = 0; c < 8; ++c) {
            Frag f;
            f.q[0] = *(const u32x4*)(qp + 32 * c + (hi ? 8 : 0));
            f.q[1] = *(const u32x4*)(qp + 32 * c + 16 + (hi ? 8 : 0));
            qf[c] = f.v;
        }
    }

    v8f zero8 = {0.f, 0.f, 0.f, 0.f, 0.f, 0.f, 0.f, 0.f};
    v8f acc[8];                       // this wave's 16 rows x 128 cols of P@V
#pragma unroll
    for (int i = 0; i < 8; ++i) acc[i] = zero8;
    float rs[8] = {};                 // partial softmax denominators

    __bf16* pw = &sP[grp][0];

    for (int j0 = 0; j0 < N_NODES; j0 += 64) {
        __syncthreads();                              // sKV free to overwrite
        {   // load K tile (64 key rows, row-major bf16) cooperatively
            const u32x4* src = (const u32x4*)(qn + (size_t)j0 * NFEAT);
            u32x4* dst = (u32x4*)sKV;
#pragma unroll
            for (int i = 0; i < 8; ++i) dst[i * 256 + tid] = src[i * 256 + tid];
        }
        __syncthreads();

        // S = Q @ K^T for this wave's 2 n-tiles; exp; partial row sums; stage P
#pragma unroll
        for (int nt = 0; nt < 2; ++nt) {
            const int ntg = half * 2 + nt;
            v8f s = zero8;
#pragma unroll
            for (int ck = 0; ck < 8; ++ck) {
                Frag b;   // B 32x16: lane = key col, K slots in table order (8h.., 16+8h..)
                const __bf16* kp = sKV + (size_t)(ntg * 16 + l16) * NFEAT
                                       + 32 * ck + (hi ? 8 : 0);
                b.q[0] = *(const u32x4*)(kp);
                b.q[1] = *(const u32x4*)(kp + 16);
                s = __builtin_amdgcn_wmma_f32_16x16x32_bf16(false, qf[ck], false, b.v,
                                                            (short)0, s, false, false);
            }
#pragma unroll
            for (int r = 0; r < 8; ++r) {
                float e = __expf(s[r]);       // logits in [-1,1]; TAU = 1
                float p = e;                  // reduce across the 16-lane half (one row)
                p += __shfl_xor(p, 1, 32);
                p += __shfl_xor(p, 2, 32);
                p += __shfl_xor(p, 4, 32);
                p += __shfl_xor(p, 8, 32);
                rs[r] += p;
                pw[(r + 8 * hi) * 72 + ntg * 16 + l16] = (__bf16)e;
            }
        }

        __syncthreads();                              // K reads done, P fully staged
        {   // load V^T tile: sKV[col][key] (pad 72 avoids bank conflicts)
#pragma unroll
            for (int it = 0; it < 8; ++it) {
                int idx = it * 256 + tid;             // 2048 chunks of 8 bf16
                int key = idx >> 5;
                int c8  = (idx & 31) * 8;
                u32x4 d = *(const u32x4*)(vv + (size_t)(j0 + key) * NFEAT + c8);
                const __bf16* e = (const __bf16*)&d;
#pragma unroll
                for (int k = 0; k < 8; ++k) sKV[(size_t)(c8 + k) * 72 + key] = e[k];
            }
        }
        __syncthreads();

        // O += P @ V over this wave's 8 column tiles (keys = 2 chunks of 32)
#pragma unroll
        for (int ct = 0; ct < 8; ++ct) {
            const int ctg = half * 8 + ct;
            v8f c = acc[ct];
#pragma unroll
            for (int ck = 0; ck < 2; ++ck) {
                Frag a, b;
                const __bf16* ap = pw + (size_t)l16 * 72 + 32 * ck + (hi ? 8 : 0);
                a.q[0] = *(const u32x4*)(ap);
                a.q[1] = *(const u32x4*)(ap + 16);
                const __bf16* bp = sKV + (size_t)(ctg * 16 + l16) * 72
                                       + 32 * ck + (hi ? 8 : 0);
                b.q[0] = *(const u32x4*)(bp);
                b.q[1] = *(const u32x4*)(bp + 16);
                c = __builtin_amdgcn_wmma_f32_16x16x32_bf16(false, a.v, false, b.v,
                                                            (short)0, c, false, false);
            }
            acc[ct] = c;
        }
    }

    // combine the pair's partial denominators
    if (l16 == 0) {
#pragma unroll
        for (int r = 0; r < 8; ++r) sRS[grp * 16 + r + 8 * hi][half] = rs[r];
    }
    __syncthreads();
    float rstot[8];
#pragma unroll
    for (int r = 0; r < 8; ++r)
        rstot[r] = sRS[grp * 16 + r + 8 * hi][0] + sRS[grp * 16 + r + 8 * hi][1];

    // out = (1+scale)*x - scale * (P@V)/rowsum   (C layout: row = r+8*hi, col = l16)
#pragma unroll
    for (int ct = 0; ct < 8; ++ct) {
        const int col = (half * 8 + ct) * 16 + l16;
#pragma unroll
        for (int r = 0; r < 8; ++r) {
            const int row = rowbase + r + 8 * hi;
            float o  = acc[ct][r] / rstot[r];
            float xv = xin[(size_t)row * NFEAT + col];
            out[(size_t)row * NFEAT + col] = (1.0f + SCALE) * xv - SCALE * o;
        }
    }
}

// ---------------------------------------------------------------------------
// Row normalize (for Q/K) + bf16 cast of raw values (for V). One block per row.
// ---------------------------------------------------------------------------
__global__ void __launch_bounds__(NFEAT)
prep_rows(const float* __restrict__ x, __bf16* __restrict__ qn, __bf16* __restrict__ vv)
{
    __shared__ float red[NFEAT];
    const int row = blockIdx.x, t = threadIdx.x;
    float v = x[(size_t)row * NFEAT + t];
    red[t] = v * v;
    __syncthreads();
    for (int s = NFEAT / 2; s > 0; s >>= 1) {
        if (t < s) red[t] += red[t + s];
        __syncthreads();
    }
    float rn = rsqrtf(red[0] + 1e-12f);
    qn[(size_t)row * NFEAT + t] = (__bf16)(v * rn);
    vv[(size_t)row * NFEAT + t] = (__bf16)v;
}

// ---------------------------------------------------------------------------
// LayerNorm over the feature dim. One block per row.
// ---------------------------------------------------------------------------
__global__ void __launch_bounds__(NFEAT)
layer_norm_k(const float* __restrict__ in, const float* __restrict__ g,
             const float* __restrict__ b, float* __restrict__ out)
{
    __shared__ float red[NFEAT];
    const int row = blockIdx.x, t = threadIdx.x;
    float v = in[(size_t)row * NFEAT + t];
    red[t] = v;
    __syncthreads();
    for (int s = NFEAT / 2; s > 0; s >>= 1) {
        if (t < s) red[t] += red[t + s];
        __syncthreads();
    }
    float mean = red[0] * (1.0f / NFEAT);
    __syncthreads();
    float d = v - mean;
    red[t] = d * d;
    __syncthreads();
    for (int s = NFEAT / 2; s > 0; s >>= 1) {
        if (t < s) red[t] += red[t + s];
        __syncthreads();
    }
    float var = red[0] * (1.0f / NFEAT);
    out[(size_t)row * NFEAT + t] = d * rsqrtf(var + 1e-5f) * g[t] + b[t];
}

__global__ void zero_k(float* __restrict__ p)
{
    p[(size_t)blockIdx.x * blockDim.x + threadIdx.x] = 0.0f;
}

// COO scatter-add: one block per edge, one thread per feature.
__global__ void __launch_bounds__(NFEAT)
spmm_k(const int* __restrict__ row, const int* __restrict__ col,
       const float* __restrict__ vals, const float* __restrict__ x,
       float* __restrict__ out)
{
    const int e = blockIdx.x, t = threadIdx.x;
    const int r = row[e], c = col[e];
    const float v = vals[e];
    atomicAdd(&out[(size_t)r * NFEAT + t], v * x[(size_t)c * NFEAT + t]);
}

// Final classifier: out[n] = h[n] @ W^T + b. One block per node.
__global__ void __launch_bounds__(NFEAT)
linear_k(const float* __restrict__ h, const float* __restrict__ W,
         const float* __restrict__ b, float* __restrict__ out)
{
    __shared__ float sh[NFEAT];
    const int n = blockIdx.x, t = threadIdx.x;
    sh[t] = h[(size_t)n * NFEAT + t];
    __syncthreads();
    if (t < NCLASS) {
        float acc = b[t];
        const float* w = W + (size_t)t * NFEAT;
        for (int k = 0; k < NFEAT; ++k) acc += sh[k] * w[k];
        out[(size_t)n * NCLASS + t] = acc;
    }
}

// ---------------------------------------------------------------------------
extern "C" void kernel_launch(void* const* d_in, const int* in_sizes, int n_in,
                              void* d_out, int out_size, void* d_ws, size_t ws_size,
                              hipStream_t stream)
{
    const float* x     = (const float*)d_in[0];
    const int*   arow  = (const int*)d_in[1];
    const int*   acol  = (const int*)d_in[2];
    const float* avals = (const float*)d_in[3];
    const float* ln0_g = (const float*)d_in[4];
    const float* ln0_b = (const float*)d_in[5];
    const float* ln_g  = (const float*)d_in[6];
    const float* ln_b  = (const float*)d_in[7];
    const float* W     = (const float*)d_in[8];
    const float* bias  = (const float*)d_in[9];
    const int n_edges  = in_sizes[1];

    const size_t HN = (size_t)N_NODES * NFEAT;
    float*  bufA = (float*)d_ws;            // 8 MB
    float*  bufB = bufA + HN;               // 8 MB
    __bf16* qn   = (__bf16*)(bufB + HN);    // 4 MB (normalized rows, bf16)
    __bf16* vvb  = qn + HN;                 // 4 MB (raw rows, bf16)

    dim3 rowsGrid(N_NODES), rowsBlk(NFEAT);
    dim3 flashGrid(N_NODES / 64), flashBlk(256);

    // h = layer_norm(contra_norm(x), ln0)
    prep_rows<<<rowsGrid, rowsBlk, 0, stream>>>(x, qn, vvb);
    contranorm_flash<<<flashGrid, flashBlk, 0, stream>>>(x, qn, vvb, bufA);
    layer_norm_k<<<rowsGrid, rowsBlk, 0, stream>>>(bufA, ln0_g, ln0_b, bufB);

    float* cur = bufB;
    float* oth = bufA;
    for (int l = 0; l < 2; ++l) {
        zero_k<<<dim3((unsigned)(HN / NFEAT)), rowsBlk, 0, stream>>>(oth);
        spmm_k<<<dim3((unsigned)n_edges), rowsBlk, 0, stream>>>(arow, acol, avals, cur, oth);
        prep_rows<<<rowsGrid, rowsBlk, 0, stream>>>(oth, qn, vvb);
        contranorm_flash<<<flashGrid, flashBlk, 0, stream>>>(oth, qn, vvb, cur);
        layer_norm_k<<<rowsGrid, rowsBlk, 0, stream>>>(cur, ln_g, ln_b, oth);
        float* t = cur; cur = oth; oth = t;
    }
    linear_k<<<rowsGrid, rowsBlk, 0, stream>>>(cur, W, bias, (float*)d_out);

    (void)n_in; (void)out_size; (void)ws_size;
}
